// MoeLayer_3856880631814
// MI455X (gfx1250) — compile-verified, matrix operand order checked by reading, not run on hardware
//
#include <hip/hip_runtime.h>
#include <hip/hip_bf16.h>

typedef __attribute__((ext_vector_type(16))) __bf16 v16bf;
typedef __attribute__((ext_vector_type(8)))  __bf16 v8bf;
typedef __attribute__((ext_vector_type(8)))  float  v8f;
typedef __attribute__((ext_vector_type(4)))  float  v4f;

#define DIM    1024
#define HIDDEN 2560
#define NEXP   4
#define NTOK   8192

#define BM 128
#define BN 128
#define BK 32
#define LDA 40   // BK + 8 bf16 padding to spread LDS banks

// Async copy: global (bf16, 32 contiguous bytes) -> LDS, no VGPR round-trip.
// INST_OFFSET is added to BOTH the LDS and global addresses (ISA 08 §4.4),
// so two b128 issues with offset 0/16 move the full 32-byte chunk.
__device__ __forceinline__ void async_copy_32B(unsigned lds_byte_addr, const void* gaddr) {
    asm volatile(
        "global_load_async_to_lds_b128 %0, %1, off\n\t"
        "global_load_async_to_lds_b128 %0, %1, off offset:16"
        :
        : "v"(lds_byte_addr), "v"(gaddr)
        : "memory");
}
__device__ __forceinline__ void wait_async_lds() {
    asm volatile("s_wait_asynccnt 0x0" ::: "memory");
}

__device__ __forceinline__ float fast_sigmoid(float v) {
    return __builtin_amdgcn_rcpf(1.0f + __expf(-v));   // v_exp_f32 + v_rcp_f32
}

// ---------------------------------------------------------------------------
// Gating: one wave32 per token. logits = x . gate_w[e], softmax->top2->renorm.
// ---------------------------------------------------------------------------
__global__ void gate_kernel(const float* __restrict__ x,
                            const float* __restrict__ gw,
                            float* __restrict__ comb) {
    const int wave = threadIdx.x >> 5;
    const int lane = threadIdx.x & 31;
    const int n = blockIdx.x * 8 + wave;
    const float* xr = x + (size_t)n * DIM;

    float acc[NEXP] = {0.f, 0.f, 0.f, 0.f};
    for (int k = lane; k < DIM; k += 32) {
        const float xv = xr[k];
        acc[0] += xv * gw[0 * DIM + k];
        acc[1] += xv * gw[1 * DIM + k];
        acc[2] += xv * gw[2 * DIM + k];
        acc[3] += xv * gw[3 * DIM + k];
    }
#pragma unroll
    for (int off = 16; off > 0; off >>= 1) {
        acc[0] += __shfl_xor(acc[0], off, 32);
        acc[1] += __shfl_xor(acc[1], off, 32);
        acc[2] += __shfl_xor(acc[2], off, 32);
        acc[3] += __shfl_xor(acc[3], off, 32);
    }
    if (lane == 0) {
        const float m = fmaxf(fmaxf(acc[0], acc[1]), fmaxf(acc[2], acc[3]));
        float p[NEXP];
#pragma unroll
        for (int e = 0; e < NEXP; e++) p[e] = __expf(acc[e] - m);
        int i0 = 0; float b0 = p[0];
#pragma unroll
        for (int e = 1; e < NEXP; e++) if (p[e] > b0) { b0 = p[e]; i0 = e; }
        int i1 = -1; float b1 = -1.f;
#pragma unroll
        for (int e = 0; e < NEXP; e++) if (e != i0 && p[e] > b1) { b1 = p[e]; i1 = e; }
        const float inv = __builtin_amdgcn_rcpf(b0 + b1);
        float c[NEXP] = {0.f, 0.f, 0.f, 0.f};
        c[i0] = b0 * inv;
        c[i1] = b1 * inv;
        *(v4f*)(comb + (size_t)n * 4) = (v4f){c[0], c[1], c[2], c[3]};
    }
}

// ---------------------------------------------------------------------------
// x f32 -> bf16, 8 elements per thread
// ---------------------------------------------------------------------------
__global__ void cvt_kernel(const float* __restrict__ x, __bf16* __restrict__ xb, int n8) {
    const int i = blockIdx.x * blockDim.x + threadIdx.x;
    if (i >= n8) return;
    const v4f* p = (const v4f*)x + (size_t)i * 2;
    const v4f a = p[0], b = p[1];
    v8bf o;
#pragma unroll
    for (int j = 0; j < 4; j++) { o[j] = (__bf16)a[j]; o[4 + j] = (__bf16)b[j]; }
    *((v8bf*)xb + i) = o;
}

// ---------------------------------------------------------------------------
// FFN stage 1: h = silu(x @ w1^T) * (x @ w3^T), per expert.
// A = xb [NTOK, DIM] bf16 (async -> LDS), B = w1/w3 [HIDDEN, DIM] f32 (cvt on stage-in)
// ---------------------------------------------------------------------------
__launch_bounds__(256, 2)
__global__ void ffn1_kernel(const __bf16* __restrict__ xb,
                            const float* __restrict__ w1,
                            const float* __restrict__ w3,
                            __bf16* __restrict__ hb) {
    __shared__ __bf16 sA[BM * LDA];
    __shared__ __bf16 sB1[BN * LDA];
    __shared__ __bf16 sB3[BN * LDA];

    const int tid  = threadIdx.x;
    const int lane = tid & 31;
    const int wave = tid >> 5;
    const int half = lane >> 4;
    const int l16  = lane & 15;
    const int wm   = wave >> 2;      // 0..1 -> 64 rows each
    const int wn   = wave & 3;       // 0..3 -> 32 cols each
    const int m0   = blockIdx.y * BM;
    const int n0   = blockIdx.x * BN;

    const int lrow = tid >> 1;           // 0..127
    const int lcol = (tid & 1) * 16;     // 0 or 16 elements

    const unsigned ldsA = (unsigned)(size_t)&sA[lrow * LDA + lcol];

    v8f acc1[4][2] = {};
    v8f acc3[4][2] = {};

    for (int k0 = 0; k0 < DIM; k0 += BK) {
        // stage A: async global->LDS (bf16 direct, no VGPR round trip)
        async_copy_32B(ldsA, xb + (size_t)(m0 + lrow) * DIM + k0 + lcol);

        // stage B1/B3 (f32 -> bf16 through VALU)
        {
            const v4f* g = (const v4f*)(w1 + (size_t)(n0 + lrow) * DIM + k0 + lcol);
            const v4f f0 = g[0], f1 = g[1], f2 = g[2], f3 = g[3];
            v8bf b0, b1;
#pragma unroll
            for (int j = 0; j < 4; j++) {
                b0[j] = (__bf16)f0[j]; b0[4 + j] = (__bf16)f1[j];
                b1[j] = (__bf16)f2[j]; b1[4 + j] = (__bf16)f3[j];
            }
            *(v8bf*)&sB1[lrow * LDA + lcol]     = b0;
            *(v8bf*)&sB1[lrow * LDA + lcol + 8] = b1;
        }
        {
            const v4f* g = (const v4f*)(w3 + (size_t)(n0 + lrow) * DIM + k0 + lcol);
            const v4f f0 = g[0], f1 = g[1], f2 = g[2], f3 = g[3];
            v8bf b0, b1;
#pragma unroll
            for (int j = 0; j < 4; j++) {
                b0[j] = (__bf16)f0[j]; b0[4 + j] = (__bf16)f1[j];
                b1[j] = (__bf16)f2[j]; b1[4 + j] = (__bf16)f3[j];
            }
            *(v8bf*)&sB3[lrow * LDA + lcol]     = b0;
            *(v8bf*)&sB3[lrow * LDA + lcol + 8] = b1;
        }
        // prefetch next K-slice of the f32 weight streams into cache
        if (k0 + BK < DIM) {
            __builtin_prefetch(w1 + (size_t)(n0 + lrow) * DIM + k0 + BK + lcol, 0, 0);
            __builtin_prefetch(w3 + (size_t)(n0 + lrow) * DIM + k0 + BK + lcol, 0, 0);
        }

        wait_async_lds();
        __syncthreads();

        v16bf af[4], bfa[2], bfb[2];
#pragma unroll
        for (int mt = 0; mt < 4; mt++) {
            const int row = wm * 64 + mt * 16 + l16;
            const v8bf lo = *(const v8bf*)&sA[row * LDA + half * 8];
            const v8bf hi = *(const v8bf*)&sA[row * LDA + 16 + half * 8];
#pragma unroll
            for (int j = 0; j < 8; j++) { af[mt][j] = lo[j]; af[mt][8 + j] = hi[j]; }
        }
#pragma unroll
        for (int nt = 0; nt < 2; nt++) {
            const int row = wn * 32 + nt * 16 + l16;
            {
                const v8bf lo = *(const v8bf*)&sB1[row * LDA + half * 8];
                const v8bf hi = *(const v8bf*)&sB1[row * LDA + 16 + half * 8];
#pragma unroll
                for (int j = 0; j < 8; j++) { bfa[nt][j] = lo[j]; bfa[nt][8 + j] = hi[j]; }
            }
            {
                const v8bf lo = *(const v8bf*)&sB3[row * LDA + half * 8];
                const v8bf hi = *(const v8bf*)&sB3[row * LDA + 16 + half * 8];
#pragma unroll
                for (int j = 0; j < 8; j++) { bfb[nt][j] = lo[j]; bfb[nt][8 + j] = hi[j]; }
            }
        }
#pragma unroll
        for (int mt = 0; mt < 4; mt++)
#pragma unroll
            for (int nt = 0; nt < 2; nt++) {
                acc1[mt][nt] = __builtin_amdgcn_wmma_f32_16x16x32_bf16(
                    false, af[mt], false, bfa[nt], (short)0, acc1[mt][nt], false, false);
                acc3[mt][nt] = __builtin_amdgcn_wmma_f32_16x16x32_bf16(
                    false, af[mt], false, bfb[nt], (short)0, acc3[mt][nt], false, false);
            }
        __syncthreads();
    }

    // epilogue: silu(h1)*h3 -> bf16 store
#pragma unroll
    for (int mt = 0; mt < 4; mt++)
#pragma unroll
        for (int nt = 0; nt < 2; nt++) {
            const int col = n0 + wn * 32 + nt * 16 + l16;
#pragma unroll
            for (int r = 0; r < 8; r++) {
                const int row = m0 + wm * 64 + mt * 16 + r + 8 * half;
                const float h1 = acc1[mt][nt][r];
                const float h3 = acc3[mt][nt][r];
                hb[(size_t)row * HIDDEN + col] = (__bf16)(h1 * fast_sigmoid(h1) * h3);
            }
        }
}

// ---------------------------------------------------------------------------
// FFN stage 2: out[n,d] += comb[n,e] * (h @ w2^T), per expert.
// A = hb [NTOK, HIDDEN] bf16 (async -> LDS), B = w2 [DIM, HIDDEN] f32
// ---------------------------------------------------------------------------
__launch_bounds__(256, 2)
__global__ void ffn2_kernel(const __bf16* __restrict__ hb,
                            const float* __restrict__ w2,
                            const float* __restrict__ comb,
                            int expert,
                            float* __restrict__ out) {
    __shared__ __bf16 sA[BM * LDA];
    __shared__ __bf16 sB[BN * LDA];

    const int tid  = threadIdx.x;
    const int lane = tid & 31;
    const int wave = tid >> 5;
    const int half = lane >> 4;
    const int l16  = lane & 15;
    const int wm   = wave >> 2;
    const int wn   = wave & 3;
    const int m0   = blockIdx.y * BM;
    const int n0   = blockIdx.x * BN;

    const int lrow = tid >> 1;
    const int lcol = (tid & 1) * 16;

    const unsigned ldsA = (unsigned)(size_t)&sA[lrow * LDA + lcol];

    v8f acc[4][2] = {};

    for (int k0 = 0; k0 < HIDDEN; k0 += BK) {
        async_copy_32B(ldsA, hb + (size_t)(m0 + lrow) * HIDDEN + k0 + lcol);
        {
            const v4f* g = (const v4f*)(w2 + (size_t)(n0 + lrow) * HIDDEN + k0 + lcol);
            const v4f f0 = g[0], f1 = g[1], f2 = g[2], f3 = g[3];
            v8bf b0, b1;
#pragma unroll
            for (int j = 0; j < 4; j++) {
                b0[j] = (__bf16)f0[j]; b0[4 + j] = (__bf16)f1[j];
                b1[j] = (__bf16)f2[j]; b1[4 + j] = (__bf16)f3[j];
            }
            *(v8bf*)&sB[lrow * LDA + lcol]     = b0;
            *(v8bf*)&sB[lrow * LDA + lcol + 8] = b1;
        }
        if (k0 + BK < HIDDEN)
            __builtin_prefetch(w2 + (size_t)(n0 + lrow) * HIDDEN + k0 + BK + lcol, 0, 0);

        wait_async_lds();
        __syncthreads();

        v16bf af[4], bff[2];
#pragma unroll
        for (int mt = 0; mt < 4; mt++) {
            const int row = wm * 64 + mt * 16 + l16;
            const v8bf lo = *(const v8bf*)&sA[row * LDA + half * 8];
            const v8bf hi = *(const v8bf*)&sA[row * LDA + 16 + half * 8];
#pragma unroll
            for (int j = 0; j < 8; j++) { af[mt][j] = lo[j]; af[mt][8 + j] = hi[j]; }
        }
#pragma unroll
        for (int nt = 0; nt < 2; nt++) {
            const int row = wn * 32 + nt * 16 + l16;
            const v8bf lo = *(const v8bf*)&sB[row * LDA + half * 8];
            const v8bf hi = *(const v8bf*)&sB[row * LDA + 16 + half * 8];
#pragma unroll
            for (int j = 0; j < 8; j++) { bff[nt][j] = lo[j]; bff[nt][8 + j] = hi[j]; }
        }
#pragma unroll
        for (int mt = 0; mt < 4; mt++)
#pragma unroll
            for (int nt = 0; nt < 2; nt++)
                acc[mt][nt] = __builtin_amdgcn_wmma_f32_16x16x32_bf16(
                    false, af[mt], false, bff[nt], (short)0, acc[mt][nt], false, false);
        __syncthreads();
    }

#pragma unroll
    for (int mt = 0; mt < 4; mt++)
#pragma unroll
        for (int nt = 0; nt < 2; nt++) {
            const int col = n0 + wn * 32 + nt * 16 + l16;
#pragma unroll
            for (int r = 0; r < 8; r++) {
                const int row = m0 + wm * 64 + mt * 16 + r + 8 * half;
                const float w = comb[(size_t)row * NEXP + expert];
                const size_t o = (size_t)row * DIM + col;
                out[o] += w * acc[mt][nt][r];
            }
        }
}

// ---------------------------------------------------------------------------
extern "C" void kernel_launch(void* const* d_in, const int* in_sizes, int n_in,
                              void* d_out, int out_size, void* d_ws, size_t ws_size,
                              hipStream_t stream) {
    const float* x  = (const float*)d_in[0];
    const float* gw = (const float*)d_in[1];
    const float* w1 = (const float*)d_in[2];
    const float* w2 = (const float*)d_in[3];
    const float* w3 = (const float*)d_in[4];
    float* out = (float*)d_out;

    char* ws = (char*)d_ws;
    float*  comb = (float*)ws;                                         // 128 KB
    __bf16* xb   = (__bf16*)(ws + (1 << 17));                          // 16 MB
    __bf16* hb   = (__bf16*)(ws + (1 << 17) + (size_t)NTOK * DIM * 2); // 40 MB

    hipMemsetAsync(d_out, 0, (size_t)out_size * sizeof(float), stream);

    gate_kernel<<<NTOK / 8, 256, 0, stream>>>(x, gw, comb);
    cvt_kernel<<<(NTOK * DIM / 8 + 255) / 256, 256, 0, stream>>>(x, xb, NTOK * DIM / 8);

    for (int e = 0; e < NEXP; e++) {
        ffn1_kernel<<<dim3(HIDDEN / BN, NTOK / BM), 256, 0, stream>>>(
            xb, w1 + (size_t)e * HIDDEN * DIM, w3 + (size_t)e * HIDDEN * DIM, hb);
        ffn2_kernel<<<dim3(DIM / BN, NTOK / BM), 256, 0, stream>>>(
            hb, w2 + (size_t)e * DIM * HIDDEN, comb, e, out);
    }
}
